// Actor_83099027243716
// MI455X (gfx1250) — compile-verified, matrix operand order checked by reading, not run on hardware
//
#include <hip/hip_runtime.h>
#include <hip/hip_bf16.h>

// ---------------------------------------------------------------------------
// Types for CDNA5 WMMA (wave32): A/B = 16 x bf16 per lane, C/D = 8 x f32.
// ---------------------------------------------------------------------------
typedef __attribute__((ext_vector_type(16))) __bf16        v16bf;
typedef __attribute__((ext_vector_type(8)))  float         v8f;
typedef __attribute__((ext_vector_type(4)))  unsigned int  v4u;

static __device__ __forceinline__ v8f wmma_bf16(v16bf a, v16bf b, v8f c) {
  // D = A(16x32 bf16) * B(32x16 bf16) + C(16x16 f32)
  return __builtin_amdgcn_wmma_f32_16x16x32_bf16(
      /*neg_a=*/false, a, /*neg_b=*/false, b,
      /*c_mod=*/(short)0, c, /*reuse_a=*/false, /*reuse_b=*/false);
}

// RNE float -> bf16 bit pattern
static __device__ __forceinline__ unsigned short f2bf(float f) {
  unsigned int u = __float_as_uint(f);
  u += 0x7FFFu + ((u >> 16) & 1u);
  return (unsigned short)(u >> 16);
}

static __device__ __forceinline__ float sigf(float x) {
  return 1.0f / (1.0f + __expf(-x));
}

// ---------------------------------------------------------------------------
// Load one 16x32 bf16 fragment (A-layout) from a row-major matrix.
//   lane 0-15 : row = lane,    K = {0..7} in v0..3,  {16..23} in v4..7
//   lane16-31 : row = lane-16, K = {8..15} in v0..3, {24..31} in v4..7
// Both K-groups are 8 contiguous bf16 = 16 bytes -> two b128 loads per lane.
// For C = A * W^T with W row-major (rowstride = K), loading W rows with this
// same layout yields the WMMA B operand (output column n = W row n).
// ---------------------------------------------------------------------------
static __device__ __forceinline__ v16bf load_frag(const unsigned short* p, int ld) {
  const int lane = threadIdx.x & 31;
  const int row  = lane & 15;
  const int kh   = (lane >> 4) << 3;           // 0 or 8
  const unsigned short* q = p + (size_t)row * ld + kh;
  union { v16bf v; v4u u[2]; } r;
  r.u[0] = *reinterpret_cast<const v4u*>(q);        // K = kh .. kh+7
  r.u[1] = *reinterpret_cast<const v4u*>(q + 16);   // K = kh+16 .. kh+23
  return r.v;
}

// ---------------------------------------------------------------------------
// f32 -> bf16 conversion kernel
// ---------------------------------------------------------------------------
__global__ void convert_f32_bf16(const float* __restrict__ in,
                                 unsigned short* __restrict__ out, int n) {
  int i = blockIdx.x * 256 + threadIdx.x;
  if (i < n) out[i] = f2bf(in[i]);
}

// ---------------------------------------------------------------------------
// Generic GEMM: C(M x N) = A(M x K) * Bw(N x K)^T + bias(N), bf16 in, f32 out.
// Block = 128 threads (4 waves); wave w owns the 32x32 tile at
//   n = (blockIdx.x*4 + w)*32 , m = blockIdx.y*32.
// Grid = (N/128, M/32). Used for the one-time xW GEMM and the small per-step
// output GEMM (N=512), where small tiles keep the grid wide.
// ---------------------------------------------------------------------------
__global__ void wmma_gemm_bias(const unsigned short* __restrict__ A,
                               const unsigned short* __restrict__ Bw,
                               const float* __restrict__ bias,
                               float* __restrict__ C,
                               int K, int ldc) {
  const int wave = threadIdx.x >> 5;
  const int m = blockIdx.y * 32;
  const int n = (blockIdx.x * 4 + wave) * 32;

  v8f acc00 = {}, acc01 = {}, acc10 = {}, acc11 = {};

  const unsigned short* A0 = A  + (size_t)m * K;
  const unsigned short* A1 = A  + (size_t)(m + 16) * K;
  const unsigned short* B0 = Bw + (size_t)n * K;
  const unsigned short* B1 = Bw + (size_t)(n + 16) * K;

  for (int kk = 0; kk < K; kk += 32) {
    v16bf a0 = load_frag(A0 + kk, K);
    v16bf a1 = load_frag(A1 + kk, K);
    v16bf b0 = load_frag(B0 + kk, K);
    v16bf b1 = load_frag(B1 + kk, K);
    acc00 = wmma_bf16(a0, b0, acc00);
    acc01 = wmma_bf16(a0, b1, acc01);
    acc10 = wmma_bf16(a1, b0, acc10);
    acc11 = wmma_bf16(a1, b1, acc11);
  }

  // C/D layout: VGPR r -> row r (lanes 0-15) or r+8 (lanes 16-31); col = lane&15
  const int lane  = threadIdx.x & 31;
  const int col   = lane & 15;
  const int rbase = (lane >> 4) * 8;
  const float bia0 = bias[n + col];
  const float bia1 = bias[n + 16 + col];
#pragma unroll
  for (int r = 0; r < 8; ++r) {
    const int row0 = m + rbase + r;
    const int row1 = row0 + 16;
    C[(size_t)row0 * ldc + n + col]      = acc00[r] + bia0;
    C[(size_t)row0 * ldc + n + 16 + col] = acc01[r] + bia1;
    C[(size_t)row1 * ldc + n + col]      = acc10[r] + bia0;
    C[(size_t)row1 * ldc + n + 16 + col] = acc11[r] + bia1;
  }
}

// ---------------------------------------------------------------------------
// Fused recurrent step: gates = xW + h*W_hh^T + b_hh  -> LSTM cell.
// Block = 128 threads (4 waves); wave g computes gate g's 32x64 tile for the
// same (batch tile m, hidden tile n): 8 accumulators fed by 2 A-fragments and
// 4 B-fragments per K-chunk (8 WMMA : 6 frag-loads, vs 4:4 for 32x32 tiles),
// so the kernel leans on the WMMA pipe instead of the vector-memory pipe.
// Gate tiles are exchanged through LDS, then all threads apply the cell,
// update c in place and write the new h (bf16). Grid = (H/64, B/32).
// h is ping-ponged across steps by the host loop.
// ---------------------------------------------------------------------------
__global__ void lstm_step_fused(const unsigned short* __restrict__ hin,   // B x H bf16
                                const unsigned short* __restrict__ Whh,   // 4H x H bf16
                                const float* __restrict__ xW,             // B x 4H f32
                                const float* __restrict__ b_hh,           // 4H
                                float* __restrict__ c,                    // B x H f32
                                unsigned short* __restrict__ hout) {      // B x H bf16
  const int H  = 1024;
  const int G4 = 4096;
  __shared__ float smem[4][32][65];   // i,f,g,o 32x64 tiles (+1 pad col)

  const int gate = threadIdx.x >> 5;
  const int m = blockIdx.y * 32;      // batch tile
  const int n = blockIdx.x * 64;      // hidden tile

  const unsigned short* A0 = hin + (size_t)m * H;
  const unsigned short* A1 = hin + (size_t)(m + 16) * H;
  const unsigned short* Wg = Whh + (size_t)(gate * H + n) * H;

  v8f acc0[4] = {v8f{}, v8f{}, v8f{}, v8f{}};   // rows m..m+15,  cols n+16j..
  v8f acc1[4] = {v8f{}, v8f{}, v8f{}, v8f{}};   // rows m+16..m+31
  for (int kk = 0; kk < H; kk += 32) {
    v16bf a0 = load_frag(A0 + kk, H);
    v16bf a1 = load_frag(A1 + kk, H);
#pragma unroll
    for (int j = 0; j < 4; ++j) {
      v16bf b = load_frag(Wg + (size_t)(16 * j) * H + kk, H);
      acc0[j] = wmma_bf16(a0, b, acc0[j]);
      acc1[j] = wmma_bf16(a1, b, acc1[j]);
    }
  }

  const int lane  = threadIdx.x & 31;
  const int col   = lane & 15;
  const int rbase = (lane >> 4) * 8;
  const int gcol  = gate * H + n;
#pragma unroll
  for (int j = 0; j < 4; ++j) {
    const int jc = 16 * j + col;
    const float bh = b_hh[gcol + jc];
#pragma unroll
    for (int r = 0; r < 8; ++r) {
      const int lr0 = rbase + r;
      const int lr1 = lr0 + 16;
      smem[gate][lr0][jc] = acc0[j][r] + xW[(size_t)(m + lr0) * G4 + gcol + jc] + bh;
      smem[gate][lr1][jc] = acc1[j][r] + xW[(size_t)(m + lr1) * G4 + gcol + jc] + bh;
    }
  }
  __syncthreads();

  // LSTM cell: 32x64 = 2048 elements / 128 threads = 16 each
  for (int e = threadIdx.x; e < 32 * 64; e += 128) {
    const int row = e >> 6;
    const int cc  = e & 63;
    const float iv = smem[0][row][cc];
    const float fv = smem[1][row][cc];
    const float gv = smem[2][row][cc];
    const float ov = smem[3][row][cc];
    const size_t idx = (size_t)(m + row) * H + n + cc;
    const float cold = c[idx];
    const float cnew = sigf(fv) * cold + sigf(iv) * tanhf(gv);
    const float hnew = sigf(ov) * tanhf(cnew);
    c[idx]    = cnew;
    hout[idx] = f2bf(hnew);
  }
}

// ---------------------------------------------------------------------------
// Host-side orchestration (graph-capture safe: only kernel launches + async copy)
// ---------------------------------------------------------------------------
extern "C" void kernel_launch(void* const* d_in, const int* in_sizes, int n_in,
                              void* d_out, int out_size, void* d_ws, size_t ws_size,
                              hipStream_t stream) {
  (void)in_sizes; (void)n_in; (void)out_size; (void)ws_size;
  const int B = 512, IN = 1024, H = 1024, N = 512, D = 64;  // matches reference

  const float* x    = (const float*)d_in[0];
  const float* h0   = (const float*)d_in[1];
  const float* c0   = (const float*)d_in[2];
  const float* Wih  = (const float*)d_in[3];
  const float* Whh  = (const float*)d_in[4];
  const float* bih  = (const float*)d_in[5];
  const float* bhh  = (const float*)d_in[6];
  const float* Wout = (const float*)d_in[7];
  const float* bout = (const float*)d_in[8];
  float* out = (float*)d_out;

  // Workspace carve-out (256B aligned); total ~30 MB
  size_t off = 0;
  auto carve = [&](size_t bytes) -> void* {
    off = (off + 255) & ~(size_t)255;
    void* p = (char*)d_ws + off;
    off += bytes;
    return p;
  };
  unsigned short* x_bf    = (unsigned short*)carve((size_t)B * IN * 2);
  unsigned short* Wih_bf  = (unsigned short*)carve((size_t)4 * H * IN * 2);
  unsigned short* Whh_bf  = (unsigned short*)carve((size_t)4 * H * H * 2);
  unsigned short* Wout_bf = (unsigned short*)carve((size_t)N * H * 2);
  float*          xW      = (float*)carve((size_t)B * 4 * H * 4);
  float*          c_ws    = (float*)carve((size_t)B * H * 4);
  unsigned short* hA      = (unsigned short*)carve((size_t)B * H * 2);
  unsigned short* hB      = (unsigned short*)carve((size_t)B * H * 2);

  // 1) one-time bf16 conversions
  auto conv = [&](const float* src, unsigned short* dst, int n) {
    convert_f32_bf16<<<(n + 255) / 256, 256, 0, stream>>>(src, dst, n);
  };
  conv(x,    x_bf,    B * IN);
  conv(Wih,  Wih_bf,  4 * H * IN);
  conv(Whh,  Whh_bf,  4 * H * H);
  conv(Wout, Wout_bf, N * H);
  conv(h0,   hA,      B * H);
  hipMemcpyAsync(c_ws, c0, (size_t)B * H * sizeof(float),
                 hipMemcpyDeviceToDevice, stream);

  // 2) xW = x @ W_ih^T + b_ih   (B x 4H)
  wmma_gemm_bias<<<dim3(4 * H / 128, B / 32), 128, 0, stream>>>(
      x_bf, Wih_bf, bih, xW, IN, 4 * H);

  // 3) recurrence: 64 dependent steps, 2 kernels each
  unsigned short* hin  = hA;
  unsigned short* hout = hB;
  for (int d = 0; d < D; ++d) {
    lstm_step_fused<<<dim3(H / 64, B / 32), 128, 0, stream>>>(
        hin, Whh_bf, xW, bhh, c_ws, hout);
    wmma_gemm_bias<<<dim3(N / 128, B / 32), 128, 0, stream>>>(
        hout, Wout_bf, bout, out + (size_t)d * B * N, H, N);
    unsigned short* t = hin; hin = hout; hout = t;
  }
}
